// RecurrentSpikingNetwork_57131654971689
// MI455X (gfx1250) — matrix operand order repaired and checked by reading, not compiled, not run on hardware
//
#include <hip/hip_runtime.h>

// ---------------------------------------------------------------------------
// Recurrent spiking network, MI455X (gfx1250, wave32, WMMA f16 16x16x32).
//
// 8 batch tiles (M=16) x 8 neuron chunks (N=128) = 64 WGs, 8 waves each,
// one 16x16 WMMA output tile per wave. W_rec chunk cached in LDS (f16,
// 256KB); z exchanged between chunk-WGs via an f16 parity-double-buffered
// ring in d_ws and staged into LDS with async global->LDS copies
// (ASYNCcnt path); x staged with fp32->f16 conversion.
// Per-tile global barrier (monotonic counter) synchronizes the 8 chunk-WGs
// each timestep. vout GEMM K-split across all 8 waves of chunk 0 with an
// LDS ds_add_f32 reduction to balance the per-step critical path.
// ---------------------------------------------------------------------------

#define NIN    700
#define NINP   704
#define NREC   1024
#define NOUT   20
#define NOUTP  32
#define BATCH  128
#define TT     500
#define NSTEPS 499

#define ALPHA_F  0.95122942450071400910f  // exp(-1/20)
#define KAPPA_F  0.95122942450071400910f
#define THRESH_F 0.1f

typedef __attribute__((ext_vector_type(16))) _Float16 v16h;
typedef __attribute__((ext_vector_type(8)))  float    v8f;
typedef __attribute__((ext_vector_type(4)))  unsigned u32x4;
typedef __attribute__((ext_vector_type(4)))  int      v4i;

union FragAB { v16h v; u32x4 q[2]; };

// ---- workspace layout (bytes) ---------------------------------------------
#define WS_BAR_OFF  0                                   // 16 x u32 barrier ctrs
#define WS_WIN_OFF  256                                 // f16 [NREC][NINP]
#define WS_WREC_OFF (WS_WIN_OFF  + NREC * NINP * 2)     // f16 [NREC][NREC]
#define WS_WOUT_OFF (WS_WREC_OFF + NREC * NREC * 2)     // f16 [NOUTP][NREC]
#define WS_ZBUF_OFF (WS_WOUT_OFF + NOUTP * NREC * 2)    // f16 [8][2][16][NREC]
#define WS_TOTAL    (WS_ZBUF_OFF + 8 * 2 * 16 * NREC * 2)  // ~3.94 MB

// ---- LDS layout (f16 elements) --------------------------------------------
#define WR_STRIDE 1032   // 1024 + 8 pad (bank stagger)
#define Z_STRIDE  1032
#define X_STRIDE  712
#define LDS_WREC_OFF 0
#define LDS_Z_OFF    (128 * WR_STRIDE)
#define LDS_X_OFF    (LDS_Z_OFF + 16 * Z_STRIDE)
#define LDS_TOTAL_H  (LDS_X_OFF + 16 * X_STRIDE)        // 160000 f16 = 320000 B
#define LDS_VSCR_OFF LDS_TOTAL_H                        // float[512] vout scratch
#define LDS_BYTES    (LDS_TOTAL_H * 2 + 512 * 4)        // 322048 B <= 327680

// ---- CDNA5 async global->LDS copy (ASYNCcnt) with safe fallback -----------
#if __has_builtin(__builtin_amdgcn_global_load_async_to_lds_b128)
#define HAVE_ASYNC_LDS 1
#else
#define HAVE_ASYNC_LDS 0
#endif

__device__ __forceinline__ void copy16_g2l(const void* gsrc, void* ldst) {
#if HAVE_ASYNC_LDS
  __builtin_amdgcn_global_load_async_to_lds_b128(
      (__attribute__((address_space(1))) v4i*)gsrc,
      (__attribute__((address_space(3))) v4i*)ldst, 0, 0);
#else
  *(u32x4*)ldst = *(const u32x4*)gsrc;
#endif
}

__device__ __forceinline__ void async_join() {
#if HAVE_ASYNC_LDS
#if __has_builtin(__builtin_amdgcn_s_wait_asynccnt)
  __builtin_amdgcn_s_wait_asynccnt(0);
#else
  asm volatile("s_wait_asynccnt 0x0" ::: "memory");
#endif
#endif
}

// ---------------------------------------------------------------------------
__device__ __forceinline__ v8f wmma_f16(v16h a, v16h b, v8f c) {
  // (neg_a, A, neg_b, B, c_mod, C, reuse_a, reuse_b)
  return __builtin_amdgcn_wmma_f32_16x16x32_f16(false, a, false, b, (short)0,
                                                c, false, false);
}

// A-fragment (16x32 f16) from an LDS tile stored row-major [16][stride].
// Lane l: M = l&15; element j: K = 16*(j>>3) + 8*(l>>4) + (j&7)
//   -> two contiguous 16B reads at k = kb + 8*(l>>4) and +16.
__device__ __forceinline__ v16h a_frag_lds(const _Float16* base, int stride,
                                           int kb, int lane) {
  const int m = lane & 15, lg = lane >> 4;
  const _Float16* p = base + m * stride + kb + lg * 8;
  FragAB f;
  f.q[0] = *(const u32x4*)(p);
  f.q[1] = *(const u32x4*)(p + 16);
  return f.v;
}

// B-fragment (32x16 f16): lane l supplies column n = l&15 with
// K = 16*(j>>3) + 8*(l>>4) + (j&7)  -> two 16B reads of one weight row (n-major store).
__device__ __forceinline__ v16h b_frag(const _Float16* wrow, int kb, int lane) {
  const _Float16* p = wrow + kb + (lane >> 4) * 8;
  FragAB f;
  f.q[0] = *(const u32x4*)(p);
  f.q[1] = *(const u32x4*)(p + 16);
  return f.v;
}

// Per-batch-tile global barrier across the 8 chunk-WGs (monotonic counter).
__device__ __forceinline__ void tile_barrier(unsigned* ctr, unsigned target) {
  __threadfence();          // make this thread's global stores device-visible
  __syncthreads();          // all threads of the WG have fenced
  if (threadIdx.x == 0) {
    __hip_atomic_fetch_add(ctr, 1u, __ATOMIC_RELEASE, __HIP_MEMORY_SCOPE_AGENT);
    while (__hip_atomic_load(ctr, __ATOMIC_ACQUIRE, __HIP_MEMORY_SCOPE_AGENT) <
           target) {
      __builtin_amdgcn_s_sleep(2);
    }
  }
  __syncthreads();
}

// ---------------------------------------------------------------------------
// Prep: convert weights fp32 -> f16 into d_ws (padded), zero barrier counters.
__global__ void snn_prep(const float* __restrict__ w_in,
                         const float* __restrict__ w_rec,
                         const float* __restrict__ w_out,
                         unsigned char* __restrict__ ws) {
  unsigned* bar   = (unsigned*)(ws + WS_BAR_OFF);
  _Float16* wInB  = (_Float16*)(ws + WS_WIN_OFF);
  _Float16* wRecB = (_Float16*)(ws + WS_WREC_OFF);
  _Float16* wOutB = (_Float16*)(ws + WS_WOUT_OFF);
  const int tid  = blockIdx.x * blockDim.x + threadIdx.x;
  const int nthr = gridDim.x * blockDim.x;
  if (tid < 16) bar[tid] = 0u;
  for (int i = tid; i < NREC * NREC; i += nthr)
    wRecB[i] = (_Float16)w_rec[i];
  for (int i = tid; i < NREC * NINP; i += nthr) {
    const int n = i / NINP, k = i - n * NINP;
    wInB[i] = (k < NIN) ? (_Float16)w_in[n * NIN + k] : (_Float16)0.0f;
  }
  for (int i = tid; i < NOUTP * NREC; i += nthr) {
    const int n = i >> 10;
    wOutB[i] = (n < NOUT) ? (_Float16)w_out[i] : (_Float16)0.0f;
  }
}

// ---------------------------------------------------------------------------
__global__ void __launch_bounds__(256, 1)
snn_main(const float* __restrict__ inputs, unsigned char* __restrict__ ws,
         float* __restrict__ out_output, float* __restrict__ out_spiking,
         float* __restrict__ out_voltage, float* __restrict__ out_vout) {
  extern __shared__ _Float16 lds[];
  _Float16* wrec_lds = lds + LDS_WREC_OFF;   // [128][WR_STRIDE]
  _Float16* z_lds    = lds + LDS_Z_OFF;      // [16][Z_STRIDE]
  _Float16* x_lds    = lds + LDS_X_OFF;      // [16][X_STRIDE]
  float*    vscr     = (float*)(lds + LDS_VSCR_OFF);  // [2][16][16]

  const _Float16* wInB  = (const _Float16*)(ws + WS_WIN_OFF);
  const _Float16* wRecB = (const _Float16*)(ws + WS_WREC_OFF);
  const _Float16* wOutB = (const _Float16*)(ws + WS_WOUT_OFF);
  _Float16* zbuf = (_Float16*)(ws + WS_ZBUF_OFF);     // [8][2][16][NREC]
  unsigned* bar  = (unsigned*)(ws + WS_BAR_OFF);

  const int tile  = blockIdx.x >> 3;   // 0..7  batch tile (16 rows)
  const int chunk = blockIdx.x & 7;    // 0..7  neuron chunk (128 n)
  const int b0 = tile * 16;
  const int n0 = chunk * 128;
  const int tid  = threadIdx.x;
  const int wave = tid >> 5;           // 0..7 -> one 16-wide N tile each
  const int lane = tid & 31;
  const int lg   = lane >> 4;
  const int nl   = lane & 15;
  const int n_loc  = wave * 16 + nl;   // 0..127 within chunk
  const int n_glob = n0 + n_loc;
  unsigned* my_bar = bar + tile;
  _Float16* ztile = zbuf + (size_t)tile * 2 * 16 * NREC;  // [2][16][NREC]

  // ---- Phase 0: async-cache W_rec chunk (f16) into LDS --------------------
  for (int i = tid; i < 128 * 128; i += 256) {
    const int r  = i >> 7;
    const int c8 = (i & 127) * 8;
    copy16_g2l(wRecB + (size_t)(n0 + r) * NREC + c8,
               wrec_lds + r * WR_STRIDE + c8);
  }
  async_join();
  // t = 0 state is zero: outputs + parity-0 z ring slice + vout scratch.
  for (int i = tid; i < 16 * 128; i += 256) {
    const int r = i >> 7, n = i & 127;
    const size_t off = ((size_t)(b0 + r) * TT) * NREC + (n0 + n);
    out_voltage[off] = 0.0f;
    out_spiking[off] = 0.0f;
    ztile[r * NREC + (n0 + n)] = (_Float16)0.0f;   // parity 0
  }
  for (int i = tid; i < 512; i += 256) vscr[i] = 0.0f;

  // ---- persistent per-thread state (WMMA C-fragment layout) ---------------
  v8f v_st     = {0.f, 0.f, 0.f, 0.f, 0.f, 0.f, 0.f, 0.f};
  v8f z_st     = {0.f, 0.f, 0.f, 0.f, 0.f, 0.f, 0.f, 0.f};
  v8f vout_st  = {0.f, 0.f, 0.f, 0.f, 0.f, 0.f, 0.f, 0.f};
  v8f vout_sum = {0.f, 0.f, 0.f, 0.f, 0.f, 0.f, 0.f, 0.f};

  unsigned epoch = 0;
  for (int t = 1; t <= NSTEPS; ++t) {
    // All 8 chunk-WGs of this tile have published z(t-1) in parity (t-1)&1.
    ++epoch;
    tile_barrier(my_bar, 8u * epoch);

    // ---- stage z(t-1) (async f16 DMA) and x(t)=inputs[:,t-1,:] ------------
    {
      const int r0 = wave * 2;  // 8 waves x 2 rows = 16 rows
      const _Float16* zsrc_base = ztile + (size_t)((t - 1) & 1) * 16 * NREC;
      for (int rr = 0; rr < 2; ++rr) {
        const int r = r0 + rr;
        const _Float16* zsrc = zsrc_base + r * NREC;
        _Float16* zdst = z_lds + r * Z_STRIDE;
#pragma unroll
        for (int i = 0; i < 4; ++i) {   // 4 x 16B per lane = 2048B per row
          const int k16 = lane * 8 + i * 256;
          copy16_g2l(zsrc + k16, zdst + k16);
        }
        const float* xsrc = inputs + ((size_t)(b0 + r) * TT + (t - 1)) * NIN;
        _Float16* xdst = x_lds + r * X_STRIDE;
        for (int k0 = lane * 4; k0 < NINP; k0 += 128) {
          if (k0 < NIN) {
            const float4 xv = *(const float4*)(xsrc + k0);
            xdst[k0 + 0] = (_Float16)xv.x;
            xdst[k0 + 1] = (_Float16)xv.y;
            xdst[k0 + 2] = (_Float16)xv.z;
            xdst[k0 + 3] = (_Float16)xv.w;
          } else {  // pad 700..703
            xdst[k0 + 0] = (_Float16)0.0f;
            xdst[k0 + 1] = (_Float16)0.0f;
            xdst[k0 + 2] = (_Float16)0.0f;
            xdst[k0 + 3] = (_Float16)0.0f;
          }
        }
      }
    }
    async_join();
    __syncthreads();

    // ---- chunk 0: v_out(t-1), K-split over all 8 waves + LDS reduction ----
    if (chunk == 0) {
      const int tn = wave & 1;        // output N-tile (0: o=0..15, 1: o=16..31)
      const int kq = wave >> 1;       // K quarter
      v8f pc = {0.f, 0.f, 0.f, 0.f, 0.f, 0.f, 0.f, 0.f};
      const _Float16* wrow = wOutB + (size_t)(tn * 16 + nl) * NREC;
#pragma unroll
      for (int kk = 0; kk < 8; ++kk) {
        const int kb = kq * 256 + kk * 32;
        const v16h a = a_frag_lds(z_lds, Z_STRIDE, kb, lane);
        const v16h b = b_frag(wrow, kb, lane);
        pc = wmma_f16(a, b, pc);
      }
      float* sc = vscr + tn * 256;
      for (int e = 0; e < 8; ++e)
        atomicAdd(&sc[(e + lg * 8) * 16 + nl], pc[e]);
      __syncthreads();
      if (wave < 2) {
        float* scf = vscr + wave * 256;
        v8f c;
        for (int e = 0; e < 8; ++e) {
          const int m = e + lg * 8;
          c[e] = KAPPA_F * vout_st[e] + scf[m * 16 + nl];
          scf[m * 16 + nl] = 0.0f;    // re-arm scratch for next step
        }
        vout_st = c;
        for (int e = 0; e < 8; ++e) vout_sum[e] += c[e];
        const int oidx = wave * 16 + nl;
        if (oidx < NOUT) {
          for (int e = 0; e < 8; ++e) {
            const int m = e + lg * 8;
            out_vout[((size_t)(b0 + m) * TT + (t - 1)) * NOUT + oidx] = c[e];
          }
        }
      }
    }

    // ---- v(t) = ALPHA*v - THRESH*z + z@Wrec^T + x@Win^T -------------------
    v8f c;
    for (int e = 0; e < 8; ++e)
      c[e] = ALPHA_F * v_st[e] - THRESH_F * z_st[e];
    {
      const _Float16* wr_row = wrec_lds + n_loc * WR_STRIDE;
#pragma unroll 4
      for (int kb = 0; kb < NREC; kb += 32) {       // 32 WMMAs (LDS weights)
        const v16h a = a_frag_lds(z_lds, Z_STRIDE, kb, lane);
        const v16h b = b_frag(wr_row, kb, lane);
        c = wmma_f16(a, b, c);
      }
      const _Float16* wi_row = wInB + (size_t)n_glob * NINP;
#pragma unroll 2
      for (int kb = 0; kb < NINP; kb += 32) {       // 22 WMMAs (L2 weights)
        const v16h a = a_frag_lds(x_lds, X_STRIDE, kb, lane);
        const v16h b = b_frag(wi_row, kb, lane);
        c = wmma_f16(a, b, c);
      }
    }
    v_st = c;
    for (int e = 0; e < 8; ++e)
      z_st[e] = (c[e] - THRESH_F) > 0.0f ? 1.0f : 0.0f;

    // ---- publish voltage(t), spiking(t) fp32 + z(t) f16 (parity t&1) ------
    {
      _Float16* zout = ztile + (size_t)(t & 1) * 16 * NREC;
      for (int e = 0; e < 8; ++e) {
        const int m = e + lg * 8;
        const size_t off = ((size_t)(b0 + m) * TT + t) * NREC + n_glob;
        out_voltage[off] = v_st[e];
        out_spiking[off] = z_st[e];
        zout[m * NREC + n_glob] = (_Float16)z_st[e];
      }
    }
  }

  // ---- epilogue: v_out(499) and time-mean output --------------------------
  ++epoch;
  tile_barrier(my_bar, 8u * epoch);
  if (chunk == 0) {
    const int r0 = wave * 2;
    const _Float16* zsrc_base = ztile + (size_t)(NSTEPS & 1) * 16 * NREC;
    for (int rr = 0; rr < 2; ++rr) {
      const int r = r0 + rr;
#pragma unroll
      for (int i = 0; i < 4; ++i) {
        const int k16 = lane * 8 + i * 256;
        copy16_g2l(zsrc_base + r * NREC + k16, z_lds + r * Z_STRIDE + k16);
      }
    }
    async_join();
    __syncthreads();
    {
      const int tn = wave & 1;
      const int kq = wave >> 1;
      v8f pc = {0.f, 0.f, 0.f, 0.f, 0.f, 0.f, 0.f, 0.f};
      const _Float16* wrow = wOutB + (size_t)(tn * 16 + nl) * NREC;
#pragma unroll
      for (int kk = 0; kk < 8; ++kk) {
        const int kb = kq * 256 + kk * 32;
        const v16h a = a_frag_lds(z_lds, Z_STRIDE, kb, lane);
        const v16h b = b_frag(wrow, kb, lane);
        pc = wmma_f16(a, b, pc);
      }
      float* sc = vscr + tn * 256;
      for (int e = 0; e < 8; ++e)
        atomicAdd(&sc[(e + lg * 8) * 16 + nl], pc[e]);
      __syncthreads();
      if (wave < 2) {
        float* scf = vscr + wave * 256;
        const int oidx = wave * 16 + nl;
        for (int e = 0; e < 8; ++e) {
          const int m = e + lg * 8;
          const float c = KAPPA_F * vout_st[e] + scf[m * 16 + nl];
          const float s = vout_sum[e] + c;
          if (oidx < NOUT) {
            out_vout[((size_t)(b0 + m) * TT + NSTEPS) * NOUT + oidx] = c;
            out_output[(size_t)(b0 + m) * NOUT + oidx] =
                s * (1.0f / (float)TT);
          }
        }
      }
    }
  }
}

// ---------------------------------------------------------------------------
extern "C" void kernel_launch(void* const* d_in, const int* in_sizes, int n_in,
                              void* d_out, int out_size, void* d_ws,
                              size_t ws_size, hipStream_t stream) {
  (void)in_sizes; (void)n_in; (void)out_size; (void)ws_size;
  const float* inputs = (const float*)d_in[0];
  const float* w_in   = (const float*)d_in[1];
  const float* w_rec  = (const float*)d_in[2];
  const float* w_out  = (const float*)d_in[3];

  float* out_output  = (float*)d_out;                              // [128][20]
  float* out_spiking = out_output + (size_t)BATCH * NOUT;          // [128][500][1024]
  float* out_voltage = out_spiking + (size_t)BATCH * TT * NREC;    // [128][500][1024]
  float* out_vout    = out_voltage + (size_t)BATCH * TT * NREC;    // [128][500][20]
  unsigned char* ws = (unsigned char*)d_ws;

  snn_prep<<<dim3(256), dim3(256), 0, stream>>>(w_in, w_rec, w_out, ws);

  (void)hipFuncSetAttribute((const void*)snn_main,
                            hipFuncAttributeMaxDynamicSharedMemorySize,
                            LDS_BYTES);
  snn_main<<<dim3(64), dim3(256), LDS_BYTES, stream>>>(
      inputs, ws, out_output, out_spiking, out_voltage, out_vout);
}
